// create_model_28716151341496
// MI455X (gfx1250) — compile-verified, matrix operand order checked by reading, not run on hardware
//
#include <hip/hip_runtime.h>
#include <hip/hip_bf16.h>

typedef __attribute__((ext_vector_type(16))) _Float16 v16h;
typedef __attribute__((ext_vector_type(8)))  float    v8f;

// ---------------------------------------------------------------------------
// WMMA helpers (wave32, 16x16x32 f16 -> f32)
// A operand layout (ISA 7.12.2): lane L<16 holds row M=L, K in {0..7,16..23};
// lane L>=16 holds row M=L-16, K shifted by +8. VGPR v packs 2 halves.
// B operand is staged in LDS as [n][k] so the identical loader applies.
// C/D layout: VGPR r, lane L -> row = r + 8*(L>=16), col = L&15.
// ---------------------------------------------------------------------------
static __device__ inline v8f wmma_f16(v16h a, v16h b, v8f c) {
  return __builtin_amdgcn_wmma_f32_16x16x32_f16(false, a, false, b, (short)0, c,
                                                false, false);
}

static __device__ inline v16h lds_load_frag(const _Float16* smem, int ld,
                                            int r0, int k0) {
  int lane = threadIdx.x & 31;
  const _Float16* p = smem + (r0 + (lane & 15)) * ld + k0 + ((lane >> 4) << 3);
  v16h v{};
#pragma unroll
  for (int i = 0; i < 4; i++) { v[2*i] = p[2*i]; v[2*i+1] = p[2*i+1]; }
#pragma unroll
  for (int i = 0; i < 4; i++) { v[8+2*i] = p[16+2*i]; v[9+2*i] = p[17+2*i]; }
  return v;
}

// Async global->LDS copy of 16 bytes per lane (CDNA5, tracked by ASYNCcnt).
static __device__ inline void async_copy_b128(uint32_t lds_off,
                                              const void* gaddr) {
  asm volatile("global_load_async_to_lds_b128 %0, %1, off"
               :: "v"(lds_off), "v"(gaddr) : "memory");
}

// ---------------------------------------------------------------------------
// Kernel 1: TposBTN = LN_N(x^T + posT); T_x = TposBTN @ Wt^T + bt
// grid = B*T blocks of 128
// ---------------------------------------------------------------------------
__global__ void k_prep_Tx(const float* __restrict__ x, const float* __restrict__ posT,
                          const float* __restrict__ tn_w, const float* __restrict__ tn_b,
                          const float* __restrict__ Wt, const float* __restrict__ bt,
                          float* __restrict__ Tx) {
  int t = blockIdx.x % 24, b = blockIdx.x / 24;
  __shared__ float z[512];
  __shared__ float rbuf[256];
  int tid = threadIdx.x;
  float vals[4];
  float lsum = 0.f, lsq = 0.f;
#pragma unroll
  for (int i = 0; i < 4; i++) {
    int nn = tid + i * 128;
    float v = x[((size_t)b * 512 + nn) * 24 + t] + posT[t * 512 + nn];
    vals[i] = v; lsum += v; lsq += v * v;
  }
  rbuf[tid] = lsum; rbuf[128 + tid] = lsq;
  __syncthreads();
  for (int s = 64; s > 0; s >>= 1) {
    if (tid < s) { rbuf[tid] += rbuf[tid + s]; rbuf[128 + tid] += rbuf[128 + tid + s]; }
    __syncthreads();
  }
  float mean = rbuf[0] * (1.f / 512.f);
  float var = rbuf[128] * (1.f / 512.f) - mean * mean;
  float rs = rsqrtf(var + 1e-5f);
#pragma unroll
  for (int i = 0; i < 4; i++) {
    int nn = tid + i * 128;
    z[nn] = (vals[i] - mean) * rs * tn_w[nn] + tn_b[nn];
  }
  __syncthreads();
  if (tid < 64) {
    int d = tid;
    float acc = bt[d];
    for (int nn = 0; nn < 512; nn++) acc += z[nn] * Wt[d * 512 + nn];
    Tx[((size_t)b * 24 + t) * 64 + d] = acc;
  }
}

// ---------------------------------------------------------------------------
// Kernel 2: Spos = LN_T(x + posS); S_x16 = (Spos @ Ws^T + bs) in f16
// grid = B*N blocks of 64
// ---------------------------------------------------------------------------
__global__ void k_prep_Sx(const float* __restrict__ x, const float* __restrict__ posS,
                          const float* __restrict__ sn_w, const float* __restrict__ sn_b,
                          const float* __restrict__ Ws, const float* __restrict__ bs,
                          _Float16* __restrict__ Sx16) {
  int n = blockIdx.x & 511, b = blockIdx.x >> 9;
  __shared__ float z[24];
  int tid = threadIdx.x;
  if (tid < 24) z[tid] = x[((size_t)b * 512 + n) * 24 + tid] + posS[n * 24 + tid];
  __syncthreads();
  float m = 0.f;
#pragma unroll
  for (int t = 0; t < 24; t++) m += z[t];
  m *= (1.f / 24.f);
  float v = 0.f;
#pragma unroll
  for (int t = 0; t < 24; t++) { float d = z[t] - m; v += d * d; }
  v *= (1.f / 24.f);
  float rs = rsqrtf(v + 1e-5f);
  int d = tid;
  float acc = bs[d];
#pragma unroll
  for (int t = 0; t < 24; t++)
    acc += ((z[t] - m) * rs * sn_w[t] + sn_b[t]) * Ws[d * 24 + t];
  Sx16[((size_t)b * 512 + n) * 64 + d] = (_Float16)acc;
}

// ---------------------------------------------------------------------------
// Kernel 3: spatial QKV projection via WMMA.
// Q[b,l,j] = sum_d Sx[b,l,d] * w[b,d,j]   (M=512,N=512,K=64 per batch/tensor)
// Q,K stored (B,H,512,64); V stored transposed (B,H,64,512) for PV B-operand.
// grid = B*3*8*8 blocks of 128 (4 waves, 64x64 output tile each)
// ---------------------------------------------------------------------------
__global__ void k_sqkv(const float* __restrict__ wq, const float* __restrict__ wk,
                       const float* __restrict__ wvv, const _Float16* __restrict__ Sx16,
                       _Float16* __restrict__ Qs, _Float16* __restrict__ Ks,
                       _Float16* __restrict__ VsT) {
  int id = blockIdx.x;
  int ntile = id & 7; id >>= 3;
  int mtile = id & 7; id >>= 3;
  int which = id % 3;
  int b = id / 3;
  const float* w = (which == 0) ? wq : (which == 1) ? wk : wvv;
  __shared__ _Float16 At[64 * 64];   // [m][k] rows of S_x
  __shared__ _Float16 Bt[64 * 64];   // [n][k] = w transposed
  int tid = threadIdx.x;
  int m0 = mtile * 64, n0 = ntile * 64;
  for (int i = tid; i < 64 * 64; i += 128) {
    int r = i >> 6, k = i & 63;
    At[i] = Sx16[((size_t)b * 512 + m0 + r) * 64 + k];
  }
  for (int i = tid; i < 64 * 64; i += 128) {
    int d = i >> 6, j = i & 63;                       // read contiguous in j
    Bt[j * 64 + d] = (_Float16)w[(size_t)b * 32768 + d * 512 + n0 + j];
  }
  __syncthreads();
  int wvid = tid >> 5;
  int mm = wvid * 16;
  v8f acc[4] = {};
  for (int ks = 0; ks < 2; ks++) {
    v16h a = lds_load_frag(At, 64, mm, ks * 32);
#pragma unroll
    for (int nt = 0; nt < 4; nt++) {
      v16h bb = lds_load_frag(Bt, 64, nt * 16, ks * 32);
      acc[nt] = wmma_f16(a, bb, acc[nt]);
    }
  }
  int lane = tid & 31;
  int nl = lane & 15, rb = (lane >> 4) ? 8 : 0;
  int h = n0 >> 6;   // n0 is a multiple of 64
  for (int nt = 0; nt < 4; nt++) {
    int dkv = nt * 16 + nl;
#pragma unroll
    for (int r = 0; r < 8; r++) {
      int l = m0 + mm + rb + r;
      _Float16 hv = (_Float16)acc[nt][r];
      if (which == 0)      Qs[(((size_t)b * 8 + h) * 512 + l) * 64 + dkv] = hv;
      else if (which == 1) Ks[(((size_t)b * 8 + h) * 512 + l) * 64 + dkv] = hv;
      else                 VsT[(((size_t)b * 8 + h) * 64 + dkv) * 512 + l] = hv;
    }
  }
}

// ---------------------------------------------------------------------------
// Kernel 4: spatial flash attention per (b,h,64-query tile).
// WMMA QK^T + PV; K/V tiles double-buffered via async global->LDS copies
// (ASYNCcnt), prefetching tile kt+1 while computing tile kt.
// mask=stam is a scalar multiplier on scores (folded into scale).
// grid = B*H*8 blocks of 128 (4 waves x 16 queries)
// ---------------------------------------------------------------------------
__global__ void k_sattn(const _Float16* __restrict__ Qs, const _Float16* __restrict__ Ks,
                        const _Float16* __restrict__ VsT, const float* __restrict__ stam,
                        _Float16* __restrict__ Oattn) {
  int id = blockIdx.x;
  int qt = id & 7; id >>= 3;
  int h = id & 7;  int b = id >> 3;
  int q0 = qt * 64;
  __shared__ _Float16 Qt[64 * 64];      // [q][dk]
  __shared__ _Float16 Kt[2][64 * 64];   // [key][dk] (B operand for QK^T)
  __shared__ _Float16 Vt[2][64 * 64];   // [dv][key] (B operand for PV)
  __shared__ _Float16 Sh[64 * 66];      // scores (f16, padded stride)
  __shared__ _Float16 Pb[64 * 64];      // [q][key] (A operand for PV)
  __shared__ float rowmax[64], rowsum[64], alpha[64];
  int tid = threadIdx.x, lane = tid & 31, wvid = tid >> 5;
  float sc = 0.125f * stam[0];
  const _Float16* Qbase = Qs + ((size_t)b * 8 + h) * 512 * 64;
  const _Float16* Kbase = Ks + ((size_t)b * 8 + h) * 512 * 64;
  const _Float16* Vbase = VsT + ((size_t)b * 8 + h) * 64 * 512;
  for (int i = tid; i < 64 * 64; i += 128) {
    int r = i >> 6, k = i & 63;
    Qt[i] = Qbase[(q0 + r) * 64 + k];
  }
  if (tid < 64) { rowmax[tid] = -1e30f; rowsum[tid] = 0.f; }

  // Issue async staging of one 64-key tile (8KB K + 8KB V) into buffer `buf`.
  // 128 threads x 4 issues x 16B per matrix = 8 async ops per thread.
  auto stage_async = [&](int buf, int k0) {
    // K tile: rows k0..k0+63 are contiguous in global (8KB block).
    const _Float16* gk = Kbase + (size_t)k0 * 64;
    uint32_t ldk = (uint32_t)(uintptr_t)&Kt[buf][0];
#pragma unroll
    for (int it = 0; it < 4; it++) {
      int idx = it * 128 + tid;                      // 0..511
      async_copy_b128(ldk + (uint32_t)idx * 16, gk + idx * 8);
    }
    // V tile: row dv (stride 512) sliced at [k0, k0+64).
    uint32_t ldv = (uint32_t)(uintptr_t)&Vt[buf][0];
#pragma unroll
    for (int it = 0; it < 4; it++) {
      int idx = it * 128 + tid;
      int dv = idx >> 3, c8 = idx & 7;
      async_copy_b128(ldv + (uint32_t)(dv * 128 + c8 * 16),
                      Vbase + (size_t)dv * 512 + k0 + c8 * 8);
    }
  };

  stage_async(0, 0);

  v8f oacc[4] = {};
  int mm = wvid * 16;
  int nl = lane & 15, rb = (lane >> 4) ? 8 : 0;
  for (int kt = 0; kt < 8; kt++) {
    int buf = kt & 1;
    if (kt < 7) {
      stage_async(buf ^ 1, (kt + 1) * 64);
      // drain the 8 oldest async ops (tile kt); async loads complete in order
      asm volatile("s_wait_asynccnt 8" ::: "memory");
    } else {
      asm volatile("s_wait_asynccnt 0" ::: "memory");
    }
    __syncthreads();
    // S = scale * Q @ K^T
    {
      v8f sacc[4] = {};
      for (int ks = 0; ks < 2; ks++) {
        v16h a = lds_load_frag(Qt, 64, mm, ks * 32);
#pragma unroll
        for (int nt = 0; nt < 4; nt++) {
          v16h bb = lds_load_frag(&Kt[buf][0], 64, nt * 16, ks * 32);
          sacc[nt] = wmma_f16(a, bb, sacc[nt]);
        }
      }
      for (int nt = 0; nt < 4; nt++)
#pragma unroll
        for (int r = 0; r < 8; r++)
          Sh[(mm + rb + r) * 66 + nt * 16 + nl] = (_Float16)(sacc[nt][r] * sc);
    }
    __syncthreads();
    // online softmax, one thread per query row
    if (tid < 64) {
      int q = tid;
      float mold = rowmax[q], tmax = -1e30f;
      for (int k = 0; k < 64; k++) tmax = fmaxf(tmax, (float)Sh[q * 66 + k]);
      float mnew = fmaxf(mold, tmax);
      float al = __expf(mold - mnew);
      float s = 0.f;
      for (int k = 0; k < 64; k++) {
        float p = __expf((float)Sh[q * 66 + k] - mnew);
        Pb[q * 64 + k] = (_Float16)p;
        s += p;
      }
      rowsum[q] = rowsum[q] * al + s;
      rowmax[q] = mnew;
      alpha[q] = al;
    }
    __syncthreads();
    // O = O*alpha + P @ V
    {
#pragma unroll
      for (int r = 0; r < 8; r++) {
        float al = alpha[mm + rb + r];
#pragma unroll
        for (int nt = 0; nt < 4; nt++) oacc[nt][r] *= al;
      }
      for (int ks = 0; ks < 2; ks++) {
        v16h a = lds_load_frag(Pb, 64, mm, ks * 32);
#pragma unroll
        for (int nt = 0; nt < 4; nt++) {
          v16h bb = lds_load_frag(&Vt[buf][0], 64, nt * 16, ks * 32);
          oacc[nt] = wmma_f16(a, bb, oacc[nt]);
        }
      }
    }
    // protect buffer `buf` from being re-staged (at kt+1) before all waves
    // finished reading it
    __syncthreads();
  }
  for (int nt = 0; nt < 4; nt++)
#pragma unroll
    for (int r = 0; r < 8; r++) {
      int q = mm + rb + r;
      float ov = oacc[nt][r] / rowsum[q];
      Oattn[((size_t)b * 512 + q0 + q) * 512 + h * 64 + nt * 16 + nl] = (_Float16)ov;
    }
}

// ---------------------------------------------------------------------------
// Kernel 5: temporal hyper-attention (L=24, tiny) per (b,h). VALU.
// grid = B*H blocks of 64
// ---------------------------------------------------------------------------
__global__ void k_tattn(const float* __restrict__ Tx, const float* __restrict__ WU,
                        const float* __restrict__ WV, const float* __restrict__ WZ,
                        float* __restrict__ tsa) {
  int h = blockIdx.x & 7, b = blockIdx.x >> 3;
  __shared__ float Txs[24 * 64];
  __shared__ float Qh[24 * 64], Kh[24 * 64], Vh[24 * 64];
  __shared__ float S[24 * 24], P[24 * 24];
  int tid = threadIdx.x;
  for (int i = tid; i < 24 * 64; i += 64) Txs[i] = Tx[(size_t)b * 24 * 64 + i];
  __syncthreads();
  {
    int d = tid;
    size_t wb = (size_t)b * 32768 + h * 64 + d;
    for (int l = 0; l < 24; l++) {
      float aq = 0.f, ak = 0.f, av = 0.f;
      for (int dd = 0; dd < 64; dd++) {
        float z = Txs[l * 64 + dd];
        aq += z * WU[wb + (size_t)dd * 512];
        ak += z * WV[wb + (size_t)dd * 512];
        av += z * WZ[wb + (size_t)dd * 512];
      }
      Qh[l * 64 + d] = aq; Kh[l * 64 + d] = ak; Vh[l * 64 + d] = av;
    }
  }
  __syncthreads();
  for (int i = tid; i < 24 * 24; i += 64) {
    int l = i / 24, j = i % 24;
    float s = 0.f;
    for (int d = 0; d < 64; d++) s += Qh[l * 64 + d] * Kh[j * 64 + d];
    S[i] = s * 0.125f;
  }
  __syncthreads();
  if (tid < 24) {
    int l = tid;
    float mx = -1e30f;
    for (int j = 0; j < 24; j++) mx = fmaxf(mx, S[l * 24 + j]);
    float s = 0.f;
    for (int j = 0; j < 24; j++) { float p = __expf(S[l * 24 + j] - mx); P[l * 24 + j] = p; s += p; }
    float inv = 1.f / s;
    for (int j = 0; j < 24; j++) P[l * 24 + j] *= inv;
  }
  __syncthreads();
  {
    int d = tid;
    for (int l = 0; l < 24; l++) {
      float a = 0.f;
      for (int j = 0; j < 24; j++) a += P[l * 24 + j] * Vh[j * 64 + d];
      tsa[((size_t)b * 24 + l) * 512 + h * 64 + d] = a;
    }
  }
}

// ---------------------------------------------------------------------------
// Kernel 6: TSAo[b,n,t] = sum_j tsa[b,t,j]*to_out_w[n,j] + to_out_b[n]
// grid = B*4 blocks of 128 (thread per n)
// ---------------------------------------------------------------------------
__global__ void k_tsao(const float* __restrict__ tsa, const float* __restrict__ Wto,
                       const float* __restrict__ bto, float* __restrict__ TSAo) {
  int chunk = blockIdx.x & 3, b = blockIdx.x >> 2;
  __shared__ float ts[24 * 512];
  int tid = threadIdx.x;
  for (int i = tid; i < 24 * 512; i += 128) ts[i] = tsa[(size_t)b * 24 * 512 + i];
  __syncthreads();
  int n = chunk * 128 + tid;
  float acc[24];
#pragma unroll
  for (int t = 0; t < 24; t++) acc[t] = 0.f;
  const float* wrow = Wto + (size_t)n * 512;
  for (int j = 0; j < 512; j++) {
    float w = wrow[j];
#pragma unroll
    for (int t = 0; t < 24; t++) acc[t] += ts[t * 512 + j] * w;
  }
  float bb = bto[n];
  size_t ob = ((size_t)b * 512 + n) * 24;
  for (int t = 0; t < 24; t++) TSAo[ob + t] = acc[t] + bb;
}

// ---------------------------------------------------------------------------
// Kernel 7: SATo[b,n,t] = sum_j Oattn[b,n,j]*sat_w[t,j] + sat_b[t]
// grid = B*4 blocks of 128 (thread per n)
// ---------------------------------------------------------------------------
__global__ void k_sato(const _Float16* __restrict__ Oattn, const float* __restrict__ satw,
                       const float* __restrict__ satb, float* __restrict__ SATo) {
  int chunk = blockIdx.x & 3, b = blockIdx.x >> 2;
  __shared__ float sw[24 * 512];
  int tid = threadIdx.x;
  for (int i = tid; i < 24 * 512; i += 128) sw[i] = satw[i];
  __syncthreads();
  int n = chunk * 128 + tid;
  const _Float16* orow = Oattn + ((size_t)b * 512 + n) * 512;
  float acc[24];
#pragma unroll
  for (int t = 0; t < 24; t++) acc[t] = 0.f;
  for (int j = 0; j < 512; j++) {
    float o = (float)orow[j];
#pragma unroll
    for (int t = 0; t < 24; t++) acc[t] += o * sw[t * 512 + j];
  }
  size_t ob = ((size_t)b * 512 + n) * 24;
  for (int t = 0; t < 24; t++) SATo[ob + t] = acc[t] + satb[t];
}

// ---------------------------------------------------------------------------
// Kernel 8: fused gate1 + gate2 + residual + final LN per (b,n).
// Both MLPs run as WMMA with t padded 24->32. Global-gate branches are exactly
// zero (LN over a singleton axis) and are elided.
// grid = B*N blocks of 128 (4 waves)
// ---------------------------------------------------------------------------
__global__ void k_gate(const float* __restrict__ x, const float* __restrict__ TSAo,
                       const float* __restrict__ SATo,
                       const float* __restrict__ l1a_w, const float* __restrict__ l1a_b,
                       const float* __restrict__ l1b_w, const float* __restrict__ l1b_b,
                       const float* __restrict__ l1n_w, const float* __restrict__ l1n_b,
                       const float* __restrict__ l2a_w, const float* __restrict__ l2a_b,
                       const float* __restrict__ l2b_w, const float* __restrict__ l2b_b,
                       const float* __restrict__ l2n_w, const float* __restrict__ l2n_b,
                       const float* __restrict__ xc_w, const float* __restrict__ xc_b,
                       const float* __restrict__ norm_w, const float* __restrict__ norm_b,
                       float* __restrict__ out) {
  int id = blockIdx.x;
  int n = id & 511, b = id >> 9;
  __shared__ _Float16 W1b[64 * 32];   // [o][c]   A operand
  __shared__ _Float16 W2a[32 * 64];   // [c2][o]  A operand
  __shared__ _Float16 W2b[64 * 32];   // [o][c2]  A operand
  __shared__ _Float16 h1T[32 * 32];   // [t][c]   B operand
  __shared__ _Float16 mxT[32 * 64];   // [t][o]   B operand
  __shared__ _Float16 h2T[32 * 32];   // [t][c2]  B operand
  __shared__ float xl1[32 * 64];      // [t][o]
  __shared__ float xl2[32 * 64];      // [t][o]
  __shared__ float tsarow[24], satrow[24], xrow[24];
  int tid = threadIdx.x, lane = tid & 31, wvid = tid >> 5;
  int nl = lane & 15, rb = (lane >> 4) ? 8 : 0;
  size_t base = ((size_t)b * 512 + n) * 24;
  if (tid < 24) {
    tsarow[tid] = TSAo[base + tid];
    satrow[tid] = SATo[base + tid];
    xrow[tid] = x[base + tid];
  }
  for (int i = tid; i < 64 * 32; i += 128) W1b[i] = (_Float16)l1b_w[i];
  for (int i = tid; i < 32 * 64; i += 128) W2a[i] = (_Float16)l2a_w[i];
  for (int i = tid; i < 64 * 32; i += 128) W2b[i] = (_Float16)l2b_w[i];
  __syncthreads();
  // layer 1a (K=2) on VALU; relu; pad t>=24 to 0
  for (int i = tid; i < 32 * 32; i += 128) {
    int t = i >> 5, c = i & 31;
    float v = 0.f;
    if (t < 24) {
      v = l1a_w[c * 2 + 0] * tsarow[t] + l1a_w[c * 2 + 1] * satrow[t] + l1a_b[c];
      v = fmaxf(v, 0.f);
    }
    h1T[i] = (_Float16)v;
  }
  __syncthreads();
  // xl1 = W1b(64x32) @ h1(32x32): 4 mtiles x 2 ntiles, K=32
  {
    int mt = wvid;
    v16h a = lds_load_frag(W1b, 32, mt * 16, 0);
    for (int nt = 0; nt < 2; nt++) {
      v16h bb = lds_load_frag(h1T, 32, nt * 16, 0);
      v8f acc = {};
      acc = wmma_f16(a, bb, acc);
#pragma unroll
      for (int r = 0; r < 8; r++) {
        int o = mt * 16 + rb + r, t = nt * 16 + nl;
        xl1[t * 64 + o] = acc[r] + l1b_b[o];
      }
    }
  }
  __syncthreads();
  // gate1: LN over t, sigmoid, mix -> mxT (f16, padded)
  if (tid < 64) {
    int o = tid;
    float m = 0.f;
    for (int t = 0; t < 24; t++) m += xl1[t * 64 + o];
    m *= (1.f / 24.f);
    float v = 0.f;
    for (int t = 0; t < 24; t++) { float d = xl1[t * 64 + o] - m; v += d * d; }
    v *= (1.f / 24.f);
    float rs = rsqrtf(v + 1e-5f);
    for (int t = 0; t < 32; t++) {
      float mv = 0.f;
      if (t < 24) {
        float nrm = (xl1[t * 64 + o] - m) * rs * l1n_w[t] + l1n_b[t];
        float g = 1.f / (1.f + __expf(-nrm));
        mv = tsarow[t] * g + satrow[t] * (1.f - g);
      }
      mxT[t * 64 + o] = (_Float16)mv;
    }
  }
  __syncthreads();
  // h2 = relu(W2a(32x64) @ mx(64x32) + b): 2 mtiles x 2 ntiles, K=64 (2 steps)
  {
    int mt = wvid & 1, nt = wvid >> 1;
    v8f acc = {};
    for (int ks = 0; ks < 2; ks++) {
      v16h a = lds_load_frag(W2a, 64, mt * 16, ks * 32);
      v16h bb = lds_load_frag(mxT, 64, nt * 16, ks * 32);
      acc = wmma_f16(a, bb, acc);
    }
#pragma unroll
    for (int r = 0; r < 8; r++) {
      int c = mt * 16 + rb + r, t = nt * 16 + nl;
      h2T[t * 32 + c] = (_Float16)fmaxf(acc[r] + l2a_b[c], 0.f);
    }
  }
  __syncthreads();
  // xl2 = W2b(64x32) @ h2(32x32)
  {
    int mt = wvid;
    v16h a = lds_load_frag(W2b, 32, mt * 16, 0);
    for (int nt = 0; nt < 2; nt++) {
      v16h bb = lds_load_frag(h2T, 32, nt * 16, 0);
      v8f acc = {};
      acc = wmma_f16(a, bb, acc);
#pragma unroll
      for (int r = 0; r < 8; r++) {
        int o = mt * 16 + rb + r, t = nt * 16 + nl;
        xl2[t * 64 + o] = acc[r] + l2b_b[o];
      }
    }
  }
  __syncthreads();
  // gate2 + residual + relu + final LN; thread per output channel o
  if (tid < 64) {
    int o = tid;
    float m = 0.f;
    for (int t = 0; t < 24; t++) m += xl2[t * 64 + o];
    m *= (1.f / 24.f);
    float v = 0.f;
    for (int t = 0; t < 24; t++) { float d = xl2[t * 64 + o] - m; v += d * d; }
    v *= (1.f / 24.f);
    float rs = rsqrtf(v + 1e-5f);
    float y[24];
    float m2 = 0.f;
    float xw = xc_w[o], xb = xc_b[o];
    for (int t = 0; t < 24; t++) {
      float nrm = (xl2[t * 64 + o] - m) * rs * l2n_w[t] + l2n_b[t];
      float g = 1.f / (1.f + __expf(-nrm));
      float mx2 = tsarow[t] * g + satrow[t] * (1.f - g);
      float yv = fmaxf(xrow[t] * xw + xb + mx2, 0.f);
      y[t] = yv; m2 += yv;
    }
    m2 *= (1.f / 24.f);
    float v2 = 0.f;
    for (int t = 0; t < 24; t++) { float d = y[t] - m2; v2 += d * d; }
    v2 *= (1.f / 24.f);
    float rs2 = rsqrtf(v2 + 1e-5f);
    size_t ob = (((size_t)b * 64 + o) * 512 + n) * 24;
    for (int t = 0; t < 24; t++)
      out[ob + t] = (y[t] - m2) * rs2 * norm_w[t] + norm_b[t];
  }
}

// ---------------------------------------------------------------------------
extern "C" void kernel_launch(void* const* d_in, const int* in_sizes, int n_in,
                              void* d_out, int out_size, void* d_ws, size_t ws_size,
                              hipStream_t stream) {
  (void)in_sizes; (void)n_in; (void)out_size; (void)ws_size;
  const float* x        = (const float*)d_in[0];
  const float* tU       = (const float*)d_in[1];
  const float* tV       = (const float*)d_in[2];
  const float* tZ       = (const float*)d_in[3];
  const float* sQw      = (const float*)d_in[4];
  const float* sKw      = (const float*)d_in[5];
  const float* sVw      = (const float*)d_in[6];
  const float* posT     = (const float*)d_in[7];
  const float* posS     = (const float*)d_in[8];
  const float* tn_w     = (const float*)d_in[9];
  const float* tn_b     = (const float*)d_in[10];
  const float* sn_w     = (const float*)d_in[11];
  const float* sn_b     = (const float*)d_in[12];
  const float* Wt       = (const float*)d_in[13];
  const float* bt       = (const float*)d_in[14];
  const float* Ws       = (const float*)d_in[15];
  const float* bs       = (const float*)d_in[16];
  const float* to_out_w = (const float*)d_in[17];
  const float* to_out_b = (const float*)d_in[18];
  const float* sat_w    = (const float*)d_in[19];
  const float* sat_b    = (const float*)d_in[20];
  const float* l1a_w    = (const float*)d_in[21];
  const float* l1a_b    = (const float*)d_in[22];
  const float* l1b_w    = (const float*)d_in[23];
  const float* l1b_b    = (const float*)d_in[24];
  const float* l1n_w    = (const float*)d_in[25];
  const float* l1n_b    = (const float*)d_in[26];
  // d_in[27..32]: g1a_w/g1a_b/g1b_w/g1b_b/g1n_w/g1n_b -> contribution is
  // exactly zero (LayerNorm over a singleton axis), skipped.
  const float* l2a_w    = (const float*)d_in[33];
  const float* l2a_b    = (const float*)d_in[34];
  const float* l2b_w    = (const float*)d_in[35];
  const float* l2b_b    = (const float*)d_in[36];
  const float* l2n_w    = (const float*)d_in[37];
  const float* l2n_b    = (const float*)d_in[38];
  // d_in[39..44]: g2* skipped for the same reason.
  const float* xc_w     = (const float*)d_in[45];
  const float* xc_b     = (const float*)d_in[46];
  const float* norm_w   = (const float*)d_in[47];
  const float* norm_b   = (const float*)d_in[48];
  const float* stam     = (const float*)d_in[49];

  char* wsp = (char*)d_ws;
  size_t off = 0;
  auto alloc = [&](size_t bytes) -> void* {
    void* p = wsp + off;
    off = (off + bytes + 255) & ~(size_t)255;
    return p;
  };
  _Float16* Sx16 = (_Float16*)alloc((size_t)32 * 512 * 64 * 2);
  _Float16* Qs   = (_Float16*)alloc((size_t)32 * 8 * 512 * 64 * 2);
  _Float16* Ks   = (_Float16*)alloc((size_t)32 * 8 * 512 * 64 * 2);
  _Float16* VsT  = (_Float16*)alloc((size_t)32 * 8 * 512 * 64 * 2);
  _Float16* Oat  = (_Float16*)alloc((size_t)32 * 512 * 512 * 2);
  float* Tx      = (float*)alloc((size_t)32 * 24 * 64 * 4);
  float* tsab    = (float*)alloc((size_t)32 * 24 * 512 * 4);
  float* TSAo    = (float*)alloc((size_t)32 * 512 * 24 * 4);
  float* SATo    = (float*)alloc((size_t)32 * 512 * 24 * 4);

  k_prep_Tx<<<32 * 24, 128, 0, stream>>>(x, posT, tn_w, tn_b, Wt, bt, Tx);
  k_prep_Sx<<<32 * 512, 64, 0, stream>>>(x, posS, sn_w, sn_b, Ws, bs, Sx16);
  k_sqkv<<<32 * 3 * 8 * 8, 128, 0, stream>>>(sQw, sKw, sVw, Sx16, Qs, Ks, VsT);
  k_sattn<<<32 * 8 * 8, 128, 0, stream>>>(Qs, Ks, VsT, stam, Oat);
  k_tattn<<<32 * 8, 64, 0, stream>>>(Tx, tU, tV, tZ, tsab);
  k_tsao<<<32 * 4, 128, 0, stream>>>(tsab, to_out_w, to_out_b, TSAo);
  k_sato<<<32 * 4, 128, 0, stream>>>(Oat, sat_w, sat_b, SATo);
  k_gate<<<32 * 512, 128, 0, stream>>>(x, TSAo, SATo,
      l1a_w, l1a_b, l1b_w, l1b_b, l1n_w, l1n_b,
      l2a_w, l2a_b, l2b_w, l2b_b, l2n_w, l2n_b,
      xc_w, xc_b, norm_w, norm_b, (float*)d_out);
}
